// SGAT_72129680769666
// MI455X (gfx1250) — matrix operand order repaired and checked by reading, not compiled, not run on hardware
//
#include <hip/hip_runtime.h>
#include <hip/hip_bf16.h>
#include <stdint.h>

// Problem dims (fixed by the reference)
#define DIMD   128
#define BATCH  16
#define NNODE  256
#define NROWS  (BATCH * NNODE)   // 4096

typedef __attribute__((ext_vector_type(16))) __bf16 v16bf;
typedef __attribute__((ext_vector_type(8)))  float  v8f;

union FragBF {
  v16bf v;
  unsigned u[8];
};

// RNE float -> bf16 (bit-level)
__device__ __forceinline__ unsigned f2bf1(float f) {
  unsigned u = __float_as_uint(f);
  return (u + 0x7FFFu + ((u >> 16) & 1u)) >> 16;
}
__device__ __forceinline__ unsigned f2bf2(float lo, float hi) {
  return f2bf1(lo) | (f2bf1(hi) << 16);
}

__device__ __forceinline__ float fast_sigmoid(float e) {
  // 1 / (1 + 2^(-e*log2(e)))  -> v_exp_f32 + v_rcp_f32
  return __builtin_amdgcn_rcpf(
      1.0f + __builtin_amdgcn_exp2f(-1.4426950408889634f * e));
}

// ---------------------------------------------------------------------------
// Kernel 0: one-pass fp32 -> bf16 conversion (4 elements / thread).
// Run once for X and once per weight matrix so the GEMM kernels never convert.
// ---------------------------------------------------------------------------
__global__ __launch_bounds__(256) void cvt_f32_to_bf16_kernel(
    const float* __restrict__ src, unsigned short* __restrict__ dst, int n) {
  const int idx = (blockIdx.x * 256 + threadIdx.x) * 4;
  if (idx < n) {
    const float4 f = *(const float4*)(src + idx);
    uint2 o;
    o.x = f2bf2(f.x, f.y);
    o.y = f2bf2(f.z, f.w);
    *(uint2*)(dst + idx) = o;
  }
}

// ---------------------------------------------------------------------------
// Kernel 1: Q = X*Wq^T + bq ; K = X*Wk^T ; Vt = (X*Wv^T)^T (bf16, [B][D][N])
// Xb [4096,128] bf16, Wb [128,128] bf16 (row e-major). One block = one 16-row
// tile of one projection; 8 waves/block, wave w owns columns [16w,16w+16).
// 4 K-steps of 32; all operand loads are contiguous b128.
// ---------------------------------------------------------------------------
__global__ __launch_bounds__(256) void sgat_proj_kernel(
    const unsigned short* __restrict__ Xb,
    const unsigned short* __restrict__ Wqb, const float* __restrict__ bq,
    const unsigned short* __restrict__ Wkb,
    const unsigned short* __restrict__ Wvb,
    float* __restrict__ Q, float* __restrict__ K,
    unsigned short* __restrict__ Vt) {
  const int rowTile = blockIdx.x;        // 0..255 over 4096 rows
  const int which   = blockIdx.y;        // 0=Q, 1=K, 2=V
  const int wave    = threadIdx.x >> 5;  // 0..7
  const int lane    = threadIdx.x & 31;
  const int half    = lane >> 4;
  const int lm      = lane & 15;
  const int colbase = wave * 16;

  const unsigned short* W = (which == 0) ? Wqb : (which == 1 ? Wkb : Wvb);
  const unsigned short* xrow = Xb + (size_t)(rowTile * 16 + lm) * DIMD;
  const unsigned short* wrow = W + (size_t)(colbase + lm) * DIMD;

  v8f c = {};
#pragma unroll
  for (int kk = 0; kk < 4; ++kk) {
    const int kb = kk * 32;
    FragBF a, b;
    // A frag (16x32 bf16): v0-3 -> k = kb+8*half+0..7 ; v4-7 -> +16
    const uint4 a0 = *(const uint4*)(xrow + kb + 8 * half);
    const uint4 a1 = *(const uint4*)(xrow + kb + 16 + 8 * half);
    a.u[0] = a0.x; a.u[1] = a0.y; a.u[2] = a0.z; a.u[3] = a0.w;
    a.u[4] = a1.x; a.u[5] = a1.y; a.u[6] = a1.z; a.u[7] = a1.w;
    // B frag (32x16 bf16): v0-3 -> k = kb+8*half+0..7 ; v4-7 -> +16
    const uint4 b0 = *(const uint4*)(wrow + kb + 8 * half);
    const uint4 b1 = *(const uint4*)(wrow + kb + 16 + 8 * half);
    b.u[0] = b0.x; b.u[1] = b0.y; b.u[2] = b0.z; b.u[3] = b0.w;
    b.u[4] = b1.x; b.u[5] = b1.y; b.u[6] = b1.z; b.u[7] = b1.w;
    c = __builtin_amdgcn_wmma_f32_16x16x32_bf16(false, a.v, false, b.v,
                                                (short)0, c, false, false);
  }

  // C/D layout: vgpr r -> row m = r + 8*half, col n = lm
  if (which == 0) {
    const float bias = bq[colbase + lm];
#pragma unroll
    for (int r = 0; r < 8; ++r) {
      const int m = r + 8 * half;
      Q[(size_t)(rowTile * 16 + m) * DIMD + colbase + lm] = c[r] + bias;
    }
  } else if (which == 1) {
#pragma unroll
    for (int r = 0; r < 8; ++r) {
      const int m = r + 8 * half;
      K[(size_t)(rowTile * 16 + m) * DIMD + colbase + lm] = c[r];
    }
  } else {
    // Store V transposed: Vt[b][d][i] with global row = b*N + i
#pragma unroll
    for (int r = 0; r < 8; ++r) {
      const int m = r + 8 * half;
      const int grow = rowTile * 16 + m;
      const int bb = grow >> 8;           // / NNODE
      const int i  = grow & (NNODE - 1);
      Vt[((size_t)bb * DIMD + colbase + lm) * NNODE + i] =
          (unsigned short)f2bf1(c[r]);
    }
  }
}

// ---------------------------------------------------------------------------
// Kernel 2: fused logits + column softmax, output P[b][j][i] (bf16).
// logit[i] = edge ? sum_d sigmoid(K[b,i,d]+Q[b,j,d])*We[d] : 0.5*sum(We).
// One block per (j, b); thread i owns source node i; float4 inner loop.
// ---------------------------------------------------------------------------
__global__ __launch_bounds__(256) void sgat_attn_kernel(
    const float* __restrict__ Q, const float* __restrict__ K,
    const int* __restrict__ A, const float* __restrict__ We,
    unsigned short* __restrict__ P) {
  __shared__ float sq[DIMD];
  __shared__ float swe[DIMD];
  __shared__ float red[256];
  const int j = blockIdx.x, b = blockIdx.y;
  const int tid = threadIdx.x;

  if (tid < DIMD) {
    sq[tid]  = Q[((size_t)b * NNODE + j) * DIMD + tid];
    swe[tid] = We[tid];
  }
  __syncthreads();

  // Non-edge constant: sigmoid(0)=0.5 for every d
  float sumWe = 0.f;
#pragma unroll 8
  for (int d = 0; d < DIMD; ++d) sumWe += swe[d];
  const float c0 = 0.5f * sumWe;

  const int i = tid;
  const int edge = A[((size_t)b * NNODE + i) * NNODE + j];
  float logit;
  if (edge == 1) {
    const float4* k4 = (const float4*)(K + ((size_t)b * NNODE + i) * DIMD);
    const float4* q4 = (const float4*)sq;
    const float4* w4 = (const float4*)swe;
    float acc = 0.f;
#pragma unroll 8
    for (int d4 = 0; d4 < DIMD / 4; ++d4) {
      const float4 kv = k4[d4];
      const float4 qv = q4[d4];
      const float4 wv = w4[d4];
      acc = fmaf(fast_sigmoid(kv.x + qv.x), wv.x, acc);
      acc = fmaf(fast_sigmoid(kv.y + qv.y), wv.y, acc);
      acc = fmaf(fast_sigmoid(kv.z + qv.z), wv.z, acc);
      acc = fmaf(fast_sigmoid(kv.w + qv.w), wv.w, acc);
    }
    logit = acc;
  } else {
    logit = c0;
  }

  // Softmax over i (256 threads): max-reduce, exp, sum-reduce
  red[tid] = logit;
  __syncthreads();
  for (int s2 = 128; s2 > 0; s2 >>= 1) {
    if (tid < s2) red[tid] = fmaxf(red[tid], red[tid + s2]);
    __syncthreads();
  }
  const float mx = red[0];
  __syncthreads();
  const float p = __builtin_amdgcn_exp2f(1.4426950408889634f * (logit - mx));
  red[tid] = p;
  __syncthreads();
  for (int s2 = 128; s2 > 0; s2 >>= 1) {
    if (tid < s2) red[tid] += red[tid + s2];
    __syncthreads();
  }
  const float inv = __builtin_amdgcn_rcpf(red[0]);
  P[((size_t)b * NNODE + j) * NNODE + i] = (unsigned short)f2bf1(p * inv);
}

// ---------------------------------------------------------------------------
// Kernel 3: out[b,j,d] = sum_i P[b][j][i] * V[b][i][d]  (per-batch GEMM,
// M=256(j), N=128(d), K=256(i)); A = P (row-major bf16), B = Vt (d-major
// bf16 so K-runs are contiguous). 8 waves/block, 8 K-steps of 32.
// ---------------------------------------------------------------------------
__global__ __launch_bounds__(256) void sgat_aggr_kernel(
    const unsigned short* __restrict__ P,   // [B][N][N] bf16 (j-major)
    const unsigned short* __restrict__ Vt,  // [B][D][N] bf16
    float* __restrict__ out) {              // [B][N][D] fp32
  const int tile = blockIdx.x;            // 0..255 over global j rows
  const int wave = threadIdx.x >> 5;      // col tile over D
  const int lane = threadIdx.x & 31;
  const int half = lane >> 4;
  const int lm   = lane & 15;
  const int colbase = wave * 16;
  const int gj = tile * 16 + lm;          // global j row (tiles never cross b)
  const int b  = tile >> 4;

  const unsigned short* prow = P + (size_t)gj * NNODE;
  const unsigned short* vrow = Vt + ((size_t)b * DIMD + colbase + lm) * NNODE;

  v8f c = {};
#pragma unroll
  for (int kk = 0; kk < 8; ++kk) {
    const int kb = kk * 32;
    FragBF a, bf;
    const uint4 a0 = *(const uint4*)(prow + kb + 8 * half);
    const uint4 a1 = *(const uint4*)(prow + kb + 16 + 8 * half);
    a.u[0] = a0.x; a.u[1] = a0.y; a.u[2] = a0.z; a.u[3] = a0.w;
    a.u[4] = a1.x; a.u[5] = a1.y; a.u[6] = a1.z; a.u[7] = a1.w;
    const uint4 b0 = *(const uint4*)(vrow + kb + 8 * half);
    const uint4 b1 = *(const uint4*)(vrow + kb + 16 + 8 * half);
    bf.u[0] = b0.x; bf.u[1] = b0.y; bf.u[2] = b0.z; bf.u[3] = b0.w;
    bf.u[4] = b1.x; bf.u[5] = b1.y; bf.u[6] = b1.z; bf.u[7] = b1.w;
    c = __builtin_amdgcn_wmma_f32_16x16x32_bf16(false, a.v, false, bf.v,
                                                (short)0, c, false, false);
  }
#pragma unroll
  for (int r = 0; r < 8; ++r) {
    const int m = r + 8 * half;
    out[(size_t)(tile * 16 + m) * DIMD + colbase + lm] = c[r];
  }
}

// ---------------------------------------------------------------------------
extern "C" void kernel_launch(void* const* d_in, const int* in_sizes, int n_in,
                              void* d_out, int out_size, void* d_ws,
                              size_t ws_size, hipStream_t stream) {
  const float* x  = (const float*)d_in[0];   // [B,N,D]
  const int*   A  = (const int*)d_in[1];     // [B,N,N]
  const float* Wq = (const float*)d_in[2];   // [D,D]
  const float* bq = (const float*)d_in[3];   // [D]
  const float* Wk = (const float*)d_in[4];   // [D,D]
  const float* Wv = (const float*)d_in[5];   // [D,D]
  const float* We = (const float*)d_in[6];   // [D,1]
  float* out = (float*)d_out;                // [B,N,D]

  // Workspace layout:
  //   Q (2MB fp32) | K (2MB fp32) | Vt (1MB bf16) | P (2MB bf16)
  //   | Xb (1MB bf16) | Wqb | Wkb | Wvb (32KB bf16 each)
  char* ws = (char*)d_ws;
  size_t off = 0;
  float* Q = (float*)(ws + off);            off += (size_t)NROWS * DIMD * 4;
  float* K = (float*)(ws + off);            off += (size_t)NROWS * DIMD * 4;
  unsigned short* Vt  = (unsigned short*)(ws + off); off += (size_t)NROWS * DIMD * 2;
  unsigned short* P   = (unsigned short*)(ws + off); off += (size_t)NROWS * NNODE * 2;
  unsigned short* Xb  = (unsigned short*)(ws + off); off += (size_t)NROWS * DIMD * 2;
  unsigned short* Wqb = (unsigned short*)(ws + off); off += (size_t)DIMD * DIMD * 2;
  unsigned short* Wkb = (unsigned short*)(ws + off); off += (size_t)DIMD * DIMD * 2;
  unsigned short* Wvb = (unsigned short*)(ws + off); off += (size_t)DIMD * DIMD * 2;

  // One-time fp32 -> bf16 conversions (X and the three weight matrices)
  {
    const int nx = NROWS * DIMD;            // 524288
    cvt_f32_to_bf16_kernel<<<nx / (256 * 4), 256, 0, stream>>>(x, Xb, nx);
    const int nw = DIMD * DIMD;             // 16384
    cvt_f32_to_bf16_kernel<<<nw / (256 * 4), 256, 0, stream>>>(Wq, Wqb, nw);
    cvt_f32_to_bf16_kernel<<<nw / (256 * 4), 256, 0, stream>>>(Wk, Wkb, nw);
    cvt_f32_to_bf16_kernel<<<nw / (256 * 4), 256, 0, stream>>>(Wv, Wvb, nw);
  }

  sgat_proj_kernel<<<dim3(NROWS / 16, 3), 256, 0, stream>>>(
      Xb, Wqb, bq, Wkb, Wvb, Q, K, Vt);
  sgat_attn_kernel<<<dim3(NNODE, BATCH), 256, 0, stream>>>(Q, K, A, We, P);
  sgat_aggr_kernel<<<dim3(NROWS / 16), 256, 0, stream>>>(P, Vt, out);
}
